// SimplifiedMLSTMCell_84980222919430
// MI455X (gfx1250) — matrix operand order repaired and checked by reading, not compiled, not run on hardware
//
#include <hip/hip_runtime.h>
#include <hip/hip_bf16.h>
#include <math.h>

// Problem constants (match reference)
#define B_   512
#define IN_  2048
#define HID_ 2048
#define H_   16
#define D_   128

typedef __attribute__((ext_vector_type(16))) __bf16 v16bf;
typedef __attribute__((ext_vector_type(8)))  __bf16 v8bf;
typedef __attribute__((ext_vector_type(8)))  float  v8f;
typedef __attribute__((ext_vector_type(4)))  float  v4f;

union V16U { v16bf v; v8bf h[2]; };

// ---------------------------------------------------------------------------
// Kernel 0: f32 -> bf16 pre-conversion (one-shot; removes all cvt VALU work
// from the GEMM inner loop so the WMMA pipe is the limiter, and halves the
// L2 fetch traffic feeding it).
// ---------------------------------------------------------------------------
__global__ __launch_bounds__(256) void cvt_bf16_kernel(
    const float* __restrict__ src, __bf16* __restrict__ dst, int n4)
{
    int i = blockIdx.x * 256 + threadIdx.x;
    const int stride = gridDim.x * 256;
    for (; i < n4; i += stride) {
        const v4f s = *(const v4f*)(src + (size_t)i * 4);
        __bf16* d = dst + (size_t)i * 4;
        d[0] = (__bf16)s.x;
        d[1] = (__bf16)s.y;
        d[2] = (__bf16)s.z;
        d[3] = (__bf16)s.w;
    }
}

// ---------------------------------------------------------------------------
// Kernel 1 (fast path): out = x @ W^T + bias on pre-converted bf16 operands.
// Wave tile 32(M) x 64(N); block = 8 waves = 64x256. blockIdx.z selects q/k/v.
// A fragment layout (16-bit A 16x32, ISA 7.12.2): lane l (row = l&15) holds
//   K = koff + {0..7} in elems 0..7 and K = koff+16+{0..7} in elems 8..15,
//   koff = (l>>4)*8.
// B fragment (32x16, col = l&15): same K order as A (koff+{0..7}, koff+16+{0..7}).
// Inner loop: 12 global_load_b128 + 8 v_wmma per K=32 step.
// ---------------------------------------------------------------------------
__global__ __launch_bounds__(256) void qkv_gemm_bf16_kernel(
    const __bf16* __restrict__ xb,
    const __bf16* __restrict__ Wqb, const float* __restrict__ bq,
    const __bf16* __restrict__ Wkb, const float* __restrict__ bk,
    const __bf16* __restrict__ Wvb, const float* __restrict__ bv,
    float* __restrict__ qo, float* __restrict__ ko, float* __restrict__ vo)
{
    const int z = blockIdx.z;
    const __bf16* __restrict__ W   = (z == 0) ? Wqb : ((z == 1) ? Wkb : Wvb);
    const float* __restrict__ bias = (z == 0) ? bq  : ((z == 1) ? bk  : bv);
    float* __restrict__ out        = (z == 0) ? qo  : ((z == 1) ? ko  : vo);

    const int t    = threadIdx.x;
    const int lane = t & 31;
    const int w    = t >> 5;
    const int wm   = w >> 2;                 // 0..1
    const int wn   = w & 3;                  // 0..3
    const int m0   = blockIdx.y * 64  + wm * 32;
    const int n0   = blockIdx.x * 256 + wn * 64;

    const int lhalf = lane >> 4;             // 0 or 1
    const int lmod  = lane & 15;

    // Per-lane base pointers (advance by 32 each K-step)
    const __bf16* apA0 = xb + (size_t)(m0 + 0  + lmod) * IN_ + lhalf * 8;
    const __bf16* apA1 = xb + (size_t)(m0 + 16 + lmod) * IN_ + lhalf * 8;
    const __bf16* bp[4];
    #pragma unroll
    for (int nj = 0; nj < 4; ++nj)
        bp[nj] = W + (size_t)(n0 + nj * 16 + lmod) * IN_ + lhalf * 8;

    v8f acc[2][4] = {};

    for (int kb2 = 0; kb2 < IN_; kb2 += 32) {
        V16U a[2];
        a[0].h[0] = *(const v8bf*)(apA0 + kb2);
        a[0].h[1] = *(const v8bf*)(apA0 + kb2 + 16);
        a[1].h[0] = *(const v8bf*)(apA1 + kb2);
        a[1].h[1] = *(const v8bf*)(apA1 + kb2 + 16);
        V16U bm[4];
        #pragma unroll
        for (int nj = 0; nj < 4; ++nj) {
            bm[nj].h[0] = *(const v8bf*)(bp[nj] + kb2);
            bm[nj].h[1] = *(const v8bf*)(bp[nj] + kb2 + 16);
        }
        #pragma unroll
        for (int mi = 0; mi < 2; ++mi)
            #pragma unroll
            for (int nj = 0; nj < 4; ++nj)
                acc[mi][nj] = __builtin_amdgcn_wmma_f32_16x16x32_bf16(
                    false, a[mi].v, false, bm[nj].v, (short)0, acc[mi][nj],
                    false, false);
    }

    // Store: C/D 16x16 f32 layout: VGPR r, lane l -> row = 8*(l>>4)+r, col = l&15
    #pragma unroll
    for (int mi = 0; mi < 2; ++mi) {
        #pragma unroll
        for (int nj = 0; nj < 4; ++nj) {
            const int row0 = m0 + mi * 16 + lhalf * 8;
            const int col  = n0 + nj * 16 + lmod;
            const float bv_ = bias[col];
            float* orow = out + (size_t)row0 * HID_ + col;
            #pragma unroll
            for (int r = 0; r < 8; ++r)
                orow[(size_t)r * HID_] = acc[mi][nj][r] + bv_;
        }
    }
}

// ---------------------------------------------------------------------------
// Kernel 1 (fallback if ws too small): converts f32->bf16 in the inner loop.
// ---------------------------------------------------------------------------
__global__ __launch_bounds__(256) void qkv_gemm_f32_kernel(
    const float* __restrict__ x,
    const float* __restrict__ Wq, const float* __restrict__ bq,
    const float* __restrict__ Wk, const float* __restrict__ bk,
    const float* __restrict__ Wv, const float* __restrict__ bv,
    float* __restrict__ qo, float* __restrict__ ko, float* __restrict__ vo)
{
    const int z = blockIdx.z;
    const float* __restrict__ W    = (z == 0) ? Wq : ((z == 1) ? Wk : Wv);
    const float* __restrict__ bias = (z == 0) ? bq : ((z == 1) ? bk : bv);
    float* __restrict__ out        = (z == 0) ? qo : ((z == 1) ? ko : vo);

    const int t    = threadIdx.x;
    const int lane = t & 31;
    const int w    = t >> 5;
    const int wm   = w >> 2;
    const int wn   = w & 3;
    const int m0   = blockIdx.y * 64  + wm * 32;
    const int n0   = blockIdx.x * 256 + wn * 64;
    const int lhalf = lane >> 4;
    const int lmod  = lane & 15;

    v8f acc[2][4] = {};

    for (int kb2 = 0; kb2 < IN_; kb2 += 32) {
        v16bf a[2];
        #pragma unroll
        for (int mi = 0; mi < 2; ++mi) {
            const float* ar = x + (size_t)(m0 + mi * 16 + lmod) * IN_ + kb2 + lhalf * 8;
            #pragma unroll
            for (int p = 0; p < 8; ++p) a[mi][p]     = (__bf16)ar[p];
            #pragma unroll
            for (int p = 0; p < 8; ++p) a[mi][8 + p] = (__bf16)ar[16 + p];
        }
        v16bf bm[4];
        #pragma unroll
        for (int nj = 0; nj < 4; ++nj) {
            const float* br = W + (size_t)(n0 + nj * 16 + lmod) * IN_ + kb2 + lhalf * 8;
            #pragma unroll
            for (int p = 0; p < 8; ++p) bm[nj][p]     = (__bf16)br[p];
            #pragma unroll
            for (int p = 0; p < 8; ++p) bm[nj][8 + p] = (__bf16)br[16 + p];
        }
        #pragma unroll
        for (int mi = 0; mi < 2; ++mi)
            #pragma unroll
            for (int nj = 0; nj < 4; ++nj)
                acc[mi][nj] = __builtin_amdgcn_wmma_f32_16x16x32_bf16(
                    false, a[mi], false, bm[nj], (short)0, acc[mi][nj], false, false);
    }

    #pragma unroll
    for (int mi = 0; mi < 2; ++mi) {
        #pragma unroll
        for (int nj = 0; nj < 4; ++nj) {
            const int row0 = m0 + mi * 16 + lhalf * 8;
            const int col  = n0 + nj * 16 + lmod;
            const float bv_ = bias[col];
            float* orow = out + (size_t)row0 * HID_ + col;
            #pragma unroll
            for (int r = 0; r < 8; ++r)
                orow[(size_t)r * HID_] = acc[mi][nj][r] + bv_;
        }
    }
}

// ---------------------------------------------------------------------------
// Kernel 2: gate pre-activations. One wave per (b,h): two 2048-dots.
// ---------------------------------------------------------------------------
__global__ __launch_bounds__(256) void gates_kernel(
    const float* __restrict__ x,
    const float* __restrict__ Wi, const float* __restrict__ bi,
    const float* __restrict__ Wf, const float* __restrict__ bf_,
    float* __restrict__ igb, float* __restrict__ fgb)
{
    const int gwave = (blockIdx.x * 256 + threadIdx.x) >> 5;
    const int lane  = threadIdx.x & 31;
    if (gwave >= B_ * H_) return;
    const int b = gwave >> 4;
    const int h = gwave & 15;
    const float* xr  = x  + (size_t)b * IN_;
    const float* wir = Wi + (size_t)h * IN_;
    const float* wfr = Wf + (size_t)h * IN_;
    float si = 0.f, sf = 0.f;
    for (int k = lane; k < IN_; k += 32) {
        const float xv = xr[k];
        si += xv * wir[k];
        sf += xv * wfr[k];
    }
    #pragma unroll
    for (int m = 16; m; m >>= 1) {
        si += __shfl_xor(si, m, 32);
        sf += __shfl_xor(sf, m, 32);
    }
    if (lane == 0) {
        igb[gwave] = __expf(si + bi[h]);
        fgb[gwave] = 1.f / (1.f + __expf(-(sf + bf_[h])));
    }
}

// ---------------------------------------------------------------------------
// Kernel 3: fused state update + matvec. One block per (b,h).
//   C_new = f*C + (i*v) k^T   -- streamed with NON-TEMPORAL b128 load/store:
//   C is 1.07 GB touched exactly once; NT keeps it from evicting the hot
//   weights/x/qkv working set from the 192 MB L2.
//   num[d] = sum_e C_new[d][e]*q[e]  (each wave covers one full row per step)
//   n_new  = f*n + i*v ; m_new = f*m + i ; den = <n_new,q> + m_new + 1e-6
// ---------------------------------------------------------------------------
__global__ __launch_bounds__(256) void update_kernel(
    const float* __restrict__ Cin, const float* __restrict__ n_in,
    const float* __restrict__ m_in,
    const float* __restrict__ qb, const float* __restrict__ kbf,
    const float* __restrict__ vb,
    const float* __restrict__ igb, const float* __restrict__ fgb,
    float* __restrict__ Cout, float* __restrict__ nout, float* __restrict__ mout,
    float* __restrict__ numb, float* __restrict__ denb)
{
    const int bh = blockIdx.x;
    const int b  = bh >> 4;
    const int h  = bh & 15;
    const int t  = threadIdx.x;

    __shared__ float sq[D_], sk[D_], siv[D_], sred[D_];

    const float ig = igb[bh];
    const float fg = fgb[bh];

    const float* qv  = qb  + (size_t)b * HID_ + h * D_;
    const float* kv  = kbf + (size_t)b * HID_ + h * D_;
    const float* vv  = vb  + (size_t)b * HID_ + h * D_;
    const float* nin = n_in + (size_t)bh * D_;
    float* nno       = nout + (size_t)bh * D_;

    if (t < D_) {
        const float q  = qv[t];
        const float iv = ig * vv[t];
        sq[t]  = q;
        sk[t]  = kv[t];
        siv[t] = iv;
        const float nn = fg * nin[t] + iv;
        nno[t]  = nn;
        sred[t] = nn * q;
    }
    __syncthreads();
    #pragma unroll
    for (int s = 64; s; s >>= 1) {
        if (t < s) sred[t] += sred[t + s];
        __syncthreads();
    }
    if (t == 0) {
        const float mn = fg * m_in[bh] + ig;
        mout[bh] = mn;
        denb[bh] = sred[0] + mn + 1e-6f;
    }

    const v4f* c4  = (const v4f*)(Cin  + (size_t)bh * D_ * D_);
    v4f*       co4 = (v4f*)      (Cout + (size_t)bh * D_ * D_);
    float* nump = numb + (size_t)bh * D_;

    #pragma unroll
    for (int j = 0; j < 16; ++j) {
        const int chunk = j * 256 + t;          // 4096 float4 chunks total
        const int d     = chunk >> 5;           // row; constant per wave
        const int off   = (chunk & 31) << 2;    // column base within row
        const v4f c     = __builtin_nontemporal_load(c4 + chunk);
        const float ivd = siv[d];
        v4f cn;
        cn.x = fg * c.x + ivd * sk[off + 0];
        cn.y = fg * c.y + ivd * sk[off + 1];
        cn.z = fg * c.z + ivd * sk[off + 2];
        cn.w = fg * c.w + ivd * sk[off + 3];
        __builtin_nontemporal_store(cn, co4 + chunk);
        float pd = cn.x * sq[off + 0] + cn.y * sq[off + 1]
                 + cn.z * sq[off + 2] + cn.w * sq[off + 3];
        #pragma unroll
        for (int m = 16; m; m >>= 1) pd += __shfl_xor(pd, m, 32);
        if ((t & 31) == 0) nump[d] = pd;        // full row dot done in-wave
    }
}

// ---------------------------------------------------------------------------
// Kernel 4: h = num/den, then LayerNorm over HID. One block per batch row.
// ---------------------------------------------------------------------------
__global__ __launch_bounds__(256) void ln_kernel(
    const float* __restrict__ numb, const float* __restrict__ denb,
    const float* __restrict__ gamma, const float* __restrict__ beta,
    float* __restrict__ hout)
{
    const int b = blockIdx.x;
    const int t = threadIdx.x;
    __shared__ float s1[256], s2[256];

    const float* nr = numb + (size_t)b * HID_;
    const float* dr = denb + (size_t)b * H_;

    float vals[8];
    float lsum = 0.f, lsq = 0.f;
    #pragma unroll
    for (int r = 0; r < 8; ++r) {
        const int j = t + r * 256;
        const float v = nr[j] / dr[j >> 7];
        vals[r] = v;
        lsum += v;
        lsq  += v * v;
    }
    s1[t] = lsum;
    s2[t] = lsq;
    __syncthreads();
    #pragma unroll
    for (int s = 128; s; s >>= 1) {
        if (t < s) { s1[t] += s1[t + s]; s2[t] += s2[t + s]; }
        __syncthreads();
    }
    const float mu  = s1[0] * (1.f / HID_);
    const float var = s2[0] * (1.f / HID_) - mu * mu;
    const float inv = rsqrtf(var + 1e-5f);

    float* orow = hout + (size_t)b * HID_;
    #pragma unroll
    for (int r = 0; r < 8; ++r) {
        const int j = t + r * 256;
        orow[j] = (vals[r] - mu) * inv * gamma[j] + beta[j];
    }
}

// ---------------------------------------------------------------------------
extern "C" void kernel_launch(void* const* d_in, const int* in_sizes, int n_in,
                              void* d_out, int out_size, void* d_ws, size_t ws_size,
                              hipStream_t stream)
{
    const float* x     = (const float*)d_in[0];
    const float* C     = (const float*)d_in[1];
    const float* n_    = (const float*)d_in[2];
    const float* m_    = (const float*)d_in[3];
    const float* Wq    = (const float*)d_in[4];
    const float* bq    = (const float*)d_in[5];
    const float* Wk    = (const float*)d_in[6];
    const float* bk    = (const float*)d_in[7];
    const float* Wv    = (const float*)d_in[8];
    const float* bv    = (const float*)d_in[9];
    const float* Wi    = (const float*)d_in[10];
    const float* bi    = (const float*)d_in[11];
    const float* Wf    = (const float*)d_in[12];
    const float* bf_   = (const float*)d_in[13];
    const float* gamma = (const float*)d_in[14];
    const float* beta  = (const float*)d_in[15];

    // Output layout: h | C_new | n_new | m_new (flat, f32)
    float* out   = (float*)d_out;
    float* h_out = out;
    float* C_out = out + (size_t)B_ * HID_;
    float* n_out = C_out + (size_t)B_ * H_ * D_ * D_;
    float* m_out = n_out + (size_t)B_ * H_ * D_;

    // Workspace layout: f32 region, then bf16 region
    float* ws   = (float*)d_ws;
    float* qb   = ws;                               // 1048576 f32
    float* kb   = qb + (size_t)B_ * HID_;           // 1048576 f32
    float* vb   = kb + (size_t)B_ * HID_;           // 1048576 f32
    float* igb  = vb + (size_t)B_ * HID_;           // 8192 f32
    float* fgb  = igb + B_ * H_;                    // 8192 f32
    float* numb = fgb + B_ * H_;                    // 1048576 f32
    float* denb = numb + (size_t)B_ * H_ * D_;      // 8192 f32
    __bf16* xb  = (__bf16*)(denb + B_ * H_);        // 1048576 bf16
    __bf16* Wqb = xb + (size_t)B_ * IN_;            // 4194304 bf16
    __bf16* Wkb = Wqb + (size_t)HID_ * IN_;
    __bf16* Wvb = Wkb + (size_t)HID_ * IN_;

    const size_t f32_elems  = 4218880;                       // through denb
    const size_t need_bytes = f32_elems * 4
                            + ((size_t)B_ * IN_ + 3 * (size_t)HID_ * IN_) * 2;

    if (ws_size >= need_bytes) {
        // Fast path: one-shot bf16 pre-conversion, then pure load+WMMA GEMM.
        cvt_bf16_kernel<<<dim3(1024), 256, 0, stream>>>(x,  xb,  (B_ * IN_) / 4);
        cvt_bf16_kernel<<<dim3(4096), 256, 0, stream>>>(Wq, Wqb, (HID_ * IN_) / 4);
        cvt_bf16_kernel<<<dim3(4096), 256, 0, stream>>>(Wk, Wkb, (HID_ * IN_) / 4);
        cvt_bf16_kernel<<<dim3(4096), 256, 0, stream>>>(Wv, Wvb, (HID_ * IN_) / 4);
        qkv_gemm_bf16_kernel<<<dim3(HID_ / 256, B_ / 64, 3), 256, 0, stream>>>(
            xb, Wqb, bq, Wkb, bk, Wvb, bv, qb, kb, vb);
    } else {
        qkv_gemm_f32_kernel<<<dim3(HID_ / 256, B_ / 64, 3), 256, 0, stream>>>(
            x, Wq, bq, Wk, bk, Wv, bv, qb, kb, vb);
    }
    gates_kernel<<<dim3((B_ * H_) / 8), 256, 0, stream>>>(
        x, Wi, bi, Wf, bf_, igb, fgb);
    update_kernel<<<dim3(B_ * H_), 256, 0, stream>>>(
        C, n_, m_, qb, kb, vb, igb, fgb, C_out, n_out, m_out, numb, denb);
    ln_kernel<<<dim3(B_), 256, 0, stream>>>(
        numb, denb, gamma, beta, h_out);
}
